// GATClassifier_19688130085111
// MI455X (gfx1250) — matrix operand order repaired and checked by reading, not compiled, not run on hardware
//
#include <hip/hip_runtime.h>
#include <math.h>

typedef __attribute__((ext_vector_type(16))) __bf16 v16bf;
typedef __attribute__((ext_vector_type(8)))  float  v8f;

#define NH 8     // heads
#define FD 128   // feature dim = NH * 16

// ---- helpers -------------------------------------------------------------

__device__ __forceinline__ __bf16 f2bf(float f) {
  unsigned u = __float_as_uint(f);
  u += 0x7FFFu + ((u >> 16) & 1u);          // round-to-nearest-even
  unsigned short s = (unsigned short)(u >> 16);
  __bf16 b;
  __builtin_memcpy(&b, &s, sizeof(b));
  return b;
}

// Load 16 bf16 (32 bytes) as two 128-bit accesses into a WMMA fragment.
__device__ __forceinline__ v16bf ld_frag(const __bf16* p, int o = 8) {
  const uint4 lo = *(const uint4*)p;
  const uint4 hi = *(const uint4*)(p + o);
  v16bf v;
  __builtin_memcpy(&v, &lo, 16);
  __builtin_memcpy(((char*)&v) + 16, &hi, 16);
  return v;
}

__device__ __forceinline__ void atomicMaxF(float* addr, float v) {
  if (v >= 0.f) atomicMax((int*)addr, __float_as_int(v));
  else          atomicMin((unsigned int*)addr, __float_as_uint(v));
}

// ---- generic small kernels ----------------------------------------------

__global__ void k_cvt_bf16(const float* __restrict__ in, __bf16* __restrict__ out, int n) {
  int i = blockIdx.x * blockDim.x + threadIdx.x;
  if (i < n) out[i] = f2bf(in[i]);
}

__global__ void k_fill_f32(float* __restrict__ p, float v, int n) {
  int i = blockIdx.x * blockDim.x + threadIdx.x;
  if (i < n) p[i] = v;
}

// ---- pack W (f32, [128][128] row-major K x N) into B-fragment layout -----
// Wfrag[(kt*8 + t)*512 + lane*16 + e] : per-lane 16 bf16 contiguous (32 B),
// so the GEMM reads each fragment as two ds_load_b128.
// e -> K mapping follows the ISA 16-bit matrix VGPR layout.

__global__ void k_pack_w(const float* __restrict__ W, __bf16* __restrict__ Wfrag) {
  int i = blockIdx.x * blockDim.x + threadIdx.x;
  if (i >= 32 * 512) return;
  const int e    = i & 15;
  const int lane = (i >> 4) & 31;
  const int frag = i >> 9;                // kt*8 + t
  const int kt = frag >> 3, t = frag & 7;
  const int hlf = lane >> 4, r = lane & 15;
  const int p = e >> 1;
  const int K = ((p >= 4) ? 16 : 0) + hlf * 8 + (p & 3) * 2 + (e & 1);
  Wfrag[i] = f2bf(W[(kt * 32 + K) * FD + t * 16 + r]);
}

// ---- WMMA GEMM: Y[nrows][128] = X[nrows][128] (bf16) @ W[128][128] -------
// Block = 256 threads = 8 waves; each wave owns TWO 16-row stripes
// (block covers 256 rows). Packed W staged in LDS (32 KB); every fragment
// load (2x ds_load_b128) feeds 2 WMMAs. 64 v_wmma per wave. Full blocks
// (all but the last) take a branch-free epilogue; only the tail block pays
// for guarded stores. EXEC stays all-ones around every WMMA.

__global__ void __launch_bounds__(256)
k_gemm128_wmma(const __bf16* __restrict__ X, const __bf16* __restrict__ Wfrag,
               float* __restrict__ Y, int nrows) {
  __shared__ __bf16 Wl[32 * 512];        // 32 fragments * 512 bf16 = 32 KB
  const int tid = threadIdx.x;
  {
    const uint4* s = (const uint4*)Wfrag;
    uint4* d = (uint4*)Wl;
    #pragma unroll
    for (int i = 0; i < 8; ++i) d[tid + 256 * i] = s[tid + 256 * i];
  }
  __syncthreads();

  const int wave = tid >> 5;
  const int lane = tid & 31;
  const int hlf  = lane >> 4;            // which 16-lane half
  const int r    = lane & 15;            // row (A) / column (B/C/D) in tile
  const long rowA = (long)blockIdx.x * 256 + wave * 32;  // stripes rowA, rowA+16
  const bool full = ((long)blockIdx.x * 256 + 256) <= nrows;  // block-uniform

  long r0 = rowA + r;
  long r1 = rowA + 16 + r;
  if (!full) {                            // clamp loads only in the tail block
    if (r0 >= nrows) r0 = nrows - 1;
    if (r1 >= nrows) r1 = nrows - 1;
  }

  // A fragments for both stripes: per lane two contiguous 16 B runs.
  v16bf a0[4], a1[4];
  #pragma unroll
  for (int kt = 0; kt < 4; ++kt) {
    a0[kt] = ld_frag(X + r0 * FD + kt * 32 + hlf * 8, 16);
    a1[kt] = ld_frag(X + r1 * FD + kt * 32 + hlf * 8, 16);
  }

  #pragma unroll
  for (int t = 0; t < 8; ++t) {          // 8 column tiles (one per head)
    v8f acc0 = {}, acc1 = {};
    #pragma unroll
    for (int kt = 0; kt < 4; ++kt) {
      const v16bf b = ld_frag(Wl + ((kt * 8 + t) * 32 + lane) * 16);
      acc0 = __builtin_amdgcn_wmma_f32_16x16x32_bf16(
          false, a0[kt], false, b, (short)0, acc0, false, false);
      acc1 = __builtin_amdgcn_wmma_f32_16x16x32_bf16(
          false, a1[kt], false, b, (short)0, acc1, false, false);
    }
    float* yp = Y + (rowA + hlf * 8) * FD + t * 16 + r;
    if (full) {                          // branch-free epilogue (195/196 blocks)
      #pragma unroll
      for (int i = 0; i < 8; ++i) {      // C/D: VGPR i -> M = i + 8*hlf, N = r
        yp[(long)i * FD]        = acc0[i];
        yp[(long)(i + 16) * FD] = acc1[i];
      }
    } else {
      #pragma unroll
      for (int i = 0; i < 8; ++i) {
        const long row = rowA + hlf * 8 + i;
        if (row < nrows)      yp[(long)i * FD]        = acc0[i];
        if (row + 16 < nrows) yp[(long)(i + 16) * FD] = acc1[i];
      }
    }
  }
}

// ---- attention scores: a1[n,h] = ft[n,h,:].al[h,:], a2 likewise ----------

__global__ void k_attn_scores(const float* __restrict__ ft,
                              const float* __restrict__ al,
                              const float* __restrict__ ar,
                              float* __restrict__ a1, float* __restrict__ a2, int n) {
  int i = blockIdx.x * blockDim.x + threadIdx.x;
  if (i >= n * NH) return;
  const int node = i >> 3, h = i & 7;
  const float4* f  = (const float4*)(ft + (long)node * FD + h * 16);
  const float4* pa = (const float4*)(al + h * 16);
  const float4* pb = (const float4*)(ar + h * 16);
  float s1 = 0.f, s2 = 0.f;
  #pragma unroll
  for (int j = 0; j < 4; ++j) {
    const float4 v = f[j], x = pa[j], y = pb[j];
    s1 += v.x * x.x + v.y * x.y + v.z * x.z + v.w * x.w;
    s2 += v.x * y.x + v.y * y.y + v.z * y.z + v.w * y.w;
  }
  a1[i] = s1; a2[i] = s2;
}

// ---- edge softmax passes (score recomputed each pass; L2-resident) -------

__device__ __forceinline__ float edge_score(const float* a1, const float* a2,
                                            int s, int d, int h) {
  const float sc = a1[s * NH + h] + a2[d * NH + h];
  return sc > 0.f ? sc : 0.2f * sc;   // LeakyReLU(0.2)
}

__global__ void k_edge_max(const int* __restrict__ src, const int* __restrict__ dst,
                           const float* __restrict__ a1, const float* __restrict__ a2,
                           float* __restrict__ emax, int ne) {
  int i = blockIdx.x * blockDim.x + threadIdx.x;
  if (i >= ne * NH) return;
  const int e = i >> 3, h = i & 7;
  const int s = src[e], d = dst[e];
  atomicMaxF(&emax[d * NH + h], edge_score(a1, a2, s, d, h));
}

__global__ void k_edge_expsum(const int* __restrict__ src, const int* __restrict__ dst,
                              const float* __restrict__ a1, const float* __restrict__ a2,
                              const float* __restrict__ emax, float* __restrict__ esum,
                              int ne) {
  int i = blockIdx.x * blockDim.x + threadIdx.x;
  if (i >= ne * NH) return;
  const int e = i >> 3, h = i & 7;
  const int s = src[e], d = dst[e];
  const float ex = __expf(edge_score(a1, a2, s, d, h) - emax[d * NH + h]);
  atomicAdd(&esum[d * NH + h], ex);
}

__global__ void k_edge_aggr(const int* __restrict__ src, const int* __restrict__ dst,
                            const float* __restrict__ a1, const float* __restrict__ a2,
                            const float* __restrict__ emax, const float* __restrict__ esum,
                            const float* __restrict__ ft, float* __restrict__ acc, int ne) {
  int i = blockIdx.x * blockDim.x + threadIdx.x;
  if (i >= ne * NH) return;
  const int e = i >> 3, h = i & 7;
  const int s = src[e], d = dst[e];
  const int dh = d * NH + h;
  const float ex  = __expf(edge_score(a1, a2, s, d, h) - emax[dh]);
  const float att = ex / esum[dh];
  const float4* f = (const float4*)(ft + (long)s * FD + h * 16);
  float* o = acc + (long)d * FD + h * 16;
  #pragma unroll
  for (int j = 0; j < 4; ++j) {
    const float4 v = f[j];
    atomicAdd(o + j * 4 + 0, v.x * att);
    atomicAdd(o + j * 4 + 1, v.y * att);
    atomicAdd(o + j * 4 + 2, v.z * att);
    atomicAdd(o + j * 4 + 3, v.w * att);
  }
}

__global__ void k_elu(float* __restrict__ p, int n) {
  int i = blockIdx.x * blockDim.x + threadIdx.x;
  if (i >= n) return;
  const float x = p[i];
  p[i] = x > 0.f ? x : __expf(x) - 1.f;
}

// ---- per-graph mean pooling ---------------------------------------------

__global__ void k_pool_cnt(const int* __restrict__ n2g, float* __restrict__ cnt, int n) {
  int i = blockIdx.x * blockDim.x + threadIdx.x;
  if (i < n) atomicAdd(&cnt[n2g[i]], 1.f);
}

__global__ void k_pool_sum(const float* __restrict__ h, const int* __restrict__ n2g,
                           float* __restrict__ hg, int n) {
  int i = blockIdx.x * blockDim.x + threadIdx.x;
  if (i >= n * 32) return;
  const int node = i >> 5, c = (i & 31) * 4;
  const int g = n2g[node];
  const float4 v = *(const float4*)(h + (long)node * FD + c);
  float* o = hg + g * FD + c;
  atomicAdd(o + 0, v.x);
  atomicAdd(o + 1, v.y);
  atomicAdd(o + 2, v.z);
  atomicAdd(o + 3, v.w);
}

// ---- tail MLP (G<=64): hg/cnt -> relu(.Wt1) -> relu(.Wt2) -> .Wt3 -------

__global__ void __launch_bounds__(256)
k_tail_mlp(const float* __restrict__ hg, const float* __restrict__ cnt,
           const float* __restrict__ Wt1, const float* __restrict__ bt1,
           const float* __restrict__ Wt2, const float* __restrict__ bt2,
           const float* __restrict__ Wt3, const float* __restrict__ bt3,
           float* __restrict__ out, int g) {
  __shared__ float hs[64 * 128];
  __shared__ float x1[64 * 64];
  __shared__ float x2[64 * 64];
  const int tid = threadIdx.x;
  for (int i = tid; i < g * 128; i += 256)
    hs[i] = hg[i] / fmaxf(cnt[i >> 7], 1.f);
  __syncthreads();
  for (int i = tid; i < g * 64; i += 256) {
    const int gg = i >> 6, j = i & 63;
    float s = bt1[j];
    for (int k = 0; k < 128; ++k) s += hs[gg * 128 + k] * Wt1[k * 64 + j];
    x1[i] = fmaxf(s, 0.f);
  }
  __syncthreads();
  for (int i = tid; i < g * 64; i += 256) {
    const int gg = i >> 6, j = i & 63;
    float s = bt2[j];
    for (int k = 0; k < 64; ++k) s += x1[gg * 64 + k] * Wt2[k * 64 + j];
    x2[i] = fmaxf(s, 0.f);
  }
  __syncthreads();
  for (int i = tid; i < g * 2; i += 256) {
    const int gg = i >> 1, j = i & 1;
    float s = bt3[j];
    for (int k = 0; k < 64; ++k) s += x2[gg * 64 + k] * Wt3[k * 2 + j];
    out[i] = s;
  }
}

// ---- host orchestration --------------------------------------------------

extern "C" void kernel_launch(void* const* d_in, const int* in_sizes, int n_in,
                              void* d_out, int out_size, void* d_ws, size_t ws_size,
                              hipStream_t stream) {
  const float* feat = (const float*)d_in[0];
  const int*   src  = (const int*)d_in[1];
  const int*   dst  = (const int*)d_in[2];
  const int*   n2g  = (const int*)d_in[3];
  const float* W1   = (const float*)d_in[4];
  const float* al1  = (const float*)d_in[5];
  const float* ar1  = (const float*)d_in[6];
  const float* W2   = (const float*)d_in[7];
  const float* al2  = (const float*)d_in[8];
  const float* ar2  = (const float*)d_in[9];
  const float* Wt1  = (const float*)d_in[10];
  const float* bt1  = (const float*)d_in[11];
  const float* Wt2  = (const float*)d_in[12];
  const float* bt2  = (const float*)d_in[13];
  const float* Wt3  = (const float*)d_in[14];
  const float* bt3  = (const float*)d_in[15];
  (void)n_in; (void)ws_size;

  const int N = in_sizes[0] / FD;
  const int E = in_sizes[1];
  const int G = out_size / 2;

  // workspace layout (256 B aligned slices)
  char* w = (char*)d_ws;
  auto take = [&](size_t bytes) -> void* {
    void* p = (void*)w;
    w += (bytes + 255) & ~(size_t)255;
    return p;
  };
  __bf16* Xb  = (__bf16*)take((size_t)N * FD * 2);
  __bf16* Wb  = (__bf16*)take((size_t)FD * FD * 2);   // packed fragments
  float*  ft  = (float*) take((size_t)N * FD * 4);
  float*  a1  = (float*) take((size_t)N * NH * 4);
  float*  a2  = (float*) take((size_t)N * NH * 4);
  float*  emx = (float*) take((size_t)N * NH * 4);
  float*  esm = (float*) take((size_t)N * NH * 4);
  float*  acc = (float*) take((size_t)N * FD * 4);
  float*  hg  = (float*) take((size_t)G * FD * 4);
  float*  cnt = (float*) take((size_t)G * 4);

  const int B = 256;
  auto cdiv = [](int a, int b) { return (a + b - 1) / b; };
  const float neg_inf = -__builtin_inff();

  auto gat_layer = [&](const float* Xf, const float* Wf,
                       const float* al, const float* ar) {
    k_cvt_bf16<<<cdiv(N * FD, B), B, 0, stream>>>(Xf, Xb, N * FD);
    k_pack_w<<<64, B, 0, stream>>>(Wf, Wb);
    k_gemm128_wmma<<<cdiv(N, 256), B, 0, stream>>>(Xb, Wb, ft, N);
    k_attn_scores<<<cdiv(N * NH, B), B, 0, stream>>>(ft, al, ar, a1, a2, N);
    k_fill_f32<<<cdiv(N * NH, B), B, 0, stream>>>(emx, neg_inf, N * NH);
    k_fill_f32<<<cdiv(N * NH, B), B, 0, stream>>>(esm, 0.f, N * NH);
    k_fill_f32<<<cdiv(N * FD, B), B, 0, stream>>>(acc, 0.f, N * FD);
    k_edge_max<<<cdiv(E * NH, B), B, 0, stream>>>(src, dst, a1, a2, emx, E);
    k_edge_expsum<<<cdiv(E * NH, B), B, 0, stream>>>(src, dst, a1, a2, emx, esm, E);
    k_edge_aggr<<<cdiv(E * NH, B), B, 0, stream>>>(src, dst, a1, a2, emx, esm, ft, acc, E);
    k_elu<<<cdiv(N * FD, B), B, 0, stream>>>(acc, N * FD);
  };

  gat_layer(feat, W1, al1, ar1);   // acc = h1
  gat_layer(acc,  W2, al2, ar2);   // Xb snapshotted before acc is re-zeroed; acc = h2

  k_fill_f32<<<cdiv(G * FD, B), B, 0, stream>>>(hg, 0.f, G * FD);
  k_fill_f32<<<1, B, 0, stream>>>(cnt, 0.f, G);
  k_pool_cnt<<<cdiv(N, B), B, 0, stream>>>(n2g, cnt, N);
  k_pool_sum<<<cdiv(N * 32, B), B, 0, stream>>>(acc, n2g, hg, N);
  k_tail_mlp<<<1, 256, 0, stream>>>(hg, cnt, Wt1, bt1, Wt2, bt2, Wt3, bt3,
                                    (float*)d_out, G);
}
